// ContextBlock_75565654606318
// MI455X (gfx1250) — matrix operand order repaired and checked by reading, not compiled, not run on hardware
//
#include <hip/hip_runtime.h>

typedef __attribute__((ext_vector_type(16))) _Float16 v16h;
typedef __attribute__((ext_vector_type(8)))  float    v8f;

#define B_DIM 8
#define T_DIM 2048
#define F_DIM 64
#define L_DIM 48
#define WAVES_PER_BLOCK 4
#define LDS_STRIDE 68   // 64 + 4 floats pad: column reads hit distinct banks

#define WMMA_F16(A, Bf, C) \
  __builtin_amdgcn_wmma_f32_16x16x32_f16(false, (A), false, (Bf), (short)0, (C), false, false)

// 16-bit A-matrix 16x32 fragment from a row-major f32 matrix.
// Lane holds row m = col; halves 0..7 -> K = k0 + 16*hf + i,
// halves 8..15 -> K = k0 + 16*hf + 8 + i.
__device__ __forceinline__ v16h a_frag_f32(const float* row_base, int ld, int k0,
                                           int col, int hf) {
  const float* r = row_base + col * ld;
  const int o1 = k0 + 16 * hf;
  const int o2 = k0 + 16 * hf + 8;
  v16h a;
#pragma unroll
  for (int i = 0; i < 8; ++i) a[i] = (_Float16)r[o1 + i];
#pragma unroll
  for (int i = 0; i < 8; ++i) a[8 + i] = (_Float16)r[o2 + i];
  return a;
}

// 16-bit B-matrix 32x16 fragment, per-lane K values contiguous in one row.
// Lane holds column n; halves h -> K = k0 + 16*hf + h.
__device__ __forceinline__ v16h b_frag_row(const float* row, int k0, int hf) {
  const int o = k0 + 16 * hf;
  v16h b;
#pragma unroll
  for (int h = 0; h < 16; ++h) b[h] = (_Float16)row[o + h];
  return b;
}

// B fragment with per-lane K values strided by ld (column of a row-major matrix).
__device__ __forceinline__ v16h b_frag_col(const float* base, int ld, int k0, int hf) {
  const int o = k0 + 16 * hf;
  v16h b;
#pragma unroll
  for (int h = 0; h < 16; ++h) b[h] = (_Float16)base[(size_t)(o + h) * ld];
  return b;
}

__global__ __launch_bounds__(32 * WAVES_PER_BLOCK, 1)
void ctx_attn_kernel(const float* __restrict__ he, const float* __restrict__ W,
                     const float* __restrict__ bias, float* __restrict__ out) {
  __shared__ float qlds[WAVES_PER_BLOCK][16][LDS_STRIDE];
  __shared__ float wlds[WAVES_PER_BLOCK][16][LDS_STRIDE];

  const int wave = threadIdx.x >> 5;
  const int lane = threadIdx.x & 31;
  const int col  = lane & 15;   // N / column index within fragment
  const int hf   = lane >> 4;   // half-wave select

  const int tile = blockIdx.x * WAVES_PER_BLOCK + wave;  // 1024 tiles total
  const int b    = tile >> 7;                            // T/16 = 128 tiles/batch
  const int t0   = (tile & 127) << 4;

  const float* heB = he + (size_t)b * T_DIM * F_DIM;

  // -------- Stage 1: Q(16x64) = he_tile(16x64) * W(64x64) + bias ------------
  const v16h ha0 = a_frag_f32(heB + (size_t)t0 * F_DIM, F_DIM, 0,  col, hf);
  const v16h ha1 = a_frag_f32(heB + (size_t)t0 * F_DIM, F_DIM, 32, col, hf);

#pragma unroll
  for (int ni = 0; ni < 4; ++ni) {
    const int n0 = ni * 16;
    const float bv = bias[n0 + col];
    v8f acc;
#pragma unroll
    for (int v = 0; v < 8; ++v) acc[v] = bv;       // bias broadcast per column
    const v16h wb0 = b_frag_col(W + n0 + col, F_DIM, 0, hf);
    acc = WMMA_F16(ha0, wb0, acc);
    const v16h wb1 = b_frag_col(W + n0 + col, F_DIM, 32, hf);
    acc = WMMA_F16(ha1, wb1, acc);
#pragma unroll
    for (int v = 0; v < 8; ++v) qlds[wave][v + 8 * hf][n0 + col] = acc[v];
  }
  // LDS ops from the same wave complete in order: our reads below see our writes.

  // -------- Stage 2: S(16x64) = Q * K^T; key slot j -> he row t0-48+j -------
  const v16h qa0 = a_frag_f32(&qlds[wave][0][0], LDS_STRIDE, 0,  col, hf);
  const v16h qa1 = a_frag_f32(&qlds[wave][0][0], LDS_STRIDE, 32, col, hf);

  v8f S[4];
#pragma unroll
  for (int ni = 0; ni < 4; ++ni) {
    const int n0 = ni * 16;
    int p = t0 - L_DIM + n0 + col;            // per-lane key row (B column j)
    p = p < 0 ? 0 : p;                        // clamped rows are masked later
    const float* krow = heB + (size_t)p * F_DIM;
    v8f acc = {0.f, 0.f, 0.f, 0.f, 0.f, 0.f, 0.f, 0.f};
    acc = WMMA_F16(qa0, b_frag_row(krow, 0,  hf), acc);
    acc = WMMA_F16(qa1, b_frag_row(krow, 32, hf), acc);
    S[ni] = acc;
  }

  // -------- Stage 3: masked softmax over key slots (band r <= j <= r+47) ----
#pragma unroll
  for (int v = 0; v < 8; ++v) {
    const int r = v + 8 * hf;                 // tile row, absolute t = t0 + r
    bool val[4];
    float mx = -3.0e38f;
#pragma unroll
    for (int ni = 0; ni < 4; ++ni) {
      const int j = ni * 16 + col;
      val[ni] = (j >= r) && (j <= r + L_DIM - 1) && (t0 + j >= L_DIM);
      if (val[ni]) mx = fmaxf(mx, S[ni][v]);
    }
#pragma unroll
    for (int d = 1; d < 16; d <<= 1) mx = fmaxf(mx, __shfl_xor(mx, d, 32));
    float e[4];
    float sum = 0.f;
#pragma unroll
    for (int ni = 0; ni < 4; ++ni) {
      e[ni] = val[ni] ? __expf(S[ni][v] - mx) : 0.f;
      sum += e[ni];
    }
#pragma unroll
    for (int d = 1; d < 16; d <<= 1) sum += __shfl_xor(sum, d, 32);
    const float inv = sum > 0.f ? 1.0f / sum : 0.f;   // empty row (t==0) -> 0
#pragma unroll
    for (int ni = 0; ni < 4; ++ni) wlds[wave][r][ni * 16 + col] = e[ni] * inv;
  }

  // -------- Stage 4: ctx(16x64) = w(16x64) * K(64x64) -----------------------
  const v16h wa0 = a_frag_f32(&wlds[wave][0][0], LDS_STRIDE, 0,  col, hf);
  const v16h wa1 = a_frag_f32(&wlds[wave][0][0], LDS_STRIDE, 32, col, hf);

#pragma unroll
  for (int ni = 0; ni < 4; ++ni) {
    const int n0 = ni * 16;
    v16h hb0, hb1;
#pragma unroll
    for (int h = 0; h < 16; ++h) {            // K = key slot j, strided rows
      int p0 = t0 - L_DIM + 16 * hf + h;       p0 = p0 < 0 ? 0 : p0;
      int p1 = t0 - L_DIM + 32 + 16 * hf + h;  p1 = p1 < 0 ? 0 : p1;
      hb0[h] = (_Float16)heB[(size_t)p0 * F_DIM + n0 + col];
      hb1[h] = (_Float16)heB[(size_t)p1 * F_DIM + n0 + col];
    }
    v8f acc = {0.f, 0.f, 0.f, 0.f, 0.f, 0.f, 0.f, 0.f};
    acc = WMMA_F16(wa0, hb0, acc);
    acc = WMMA_F16(wa1, hb1, acc);
    if (t0 == 0 && hf == 0) acc[0] = heB[n0 + col];   // t==0: ctx = he[b,0,:]
#pragma unroll
    for (int v = 0; v < 8; ++v)
      out[((size_t)b * T_DIM + t0 + v + 8 * hf) * F_DIM + n0 + col] = acc[v];
  }
}

extern "C" void kernel_launch(void* const* d_in, const int* in_sizes, int n_in,
                              void* d_out, int out_size, void* d_ws, size_t ws_size,
                              hipStream_t stream) {
  const float* he   = (const float*)d_in[0];
  const float* W    = (const float*)d_in[1];
  const float* bias = (const float*)d_in[2];
  float* out        = (float*)d_out;

  const int tiles = B_DIM * (T_DIM / 16);        // 1024 wave-tiles
  dim3 grid(tiles / WAVES_PER_BLOCK);            // 256 blocks
  dim3 block(32 * WAVES_PER_BLOCK);              // 128 threads = 4 waves
  ctx_attn_kernel<<<grid, block, 0, stream>>>(he, W, bias, out);
}
